// Attention_53687091200195
// MI455X (gfx1250) — matrix operand order repaired and checked by reading, not compiled, not run on hardware
//
#include <hip/hip_runtime.h>

typedef __attribute__((ext_vector_type(16))) _Float16 v16h;
typedef __attribute__((ext_vector_type(8)))  _Float16 v8h;
typedef __attribute__((ext_vector_type(8)))  float    v8f;
typedef __attribute__((ext_vector_type(4)))  float    v4f;

#define B_   16
#define N_   2048
#define CIN  64
#define D_   128

static __device__ __forceinline__ v16h cat8(v8h lo, v8h hi) {
  return __builtin_shufflevector(lo, hi, 0,1,2,3,4,5,6,7,8,9,10,11,12,13,14,15);
}

// ---------------------------------------------------------------------------
// Kernel 0: Wt[m][d][c] = (f16) W_m[c][d]   (m: 0=Wq, 1=Wk, 2=Wv)
// Makes WMMA B-fragments of the weights contiguous 32B loads.
// ---------------------------------------------------------------------------
__global__ __launch_bounds__(256) void wt_kernel(
    const float* __restrict__ Wq, const float* __restrict__ Wk,
    const float* __restrict__ Wv, _Float16* __restrict__ Wt) {
  int idx = blockIdx.x * 256 + threadIdx.x;       // 3*128*64 = 24576 exact
  int m = idx / (D_ * CIN);
  int r = idx % (D_ * CIN);
  int d = r / CIN, c = r % CIN;
  const float* W = (m == 0) ? Wq : ((m == 1) ? Wk : Wv);
  Wt[idx] = (_Float16)W[c * D_ + d];
}

// ---------------------------------------------------------------------------
// Kernel 1: QKV projection. grid.x = B*N/16 row tiles, 8 waves = 8 D-tiles.
// Q gets the 1/sqrt(64)=0.125 scale folded in. V is stored transposed.
// ---------------------------------------------------------------------------
__global__ __launch_bounds__(256) void qkv_kernel(
    const float* __restrict__ x,
    const float* __restrict__ bq, const float* __restrict__ bk,
    const float* __restrict__ bv,
    const _Float16* __restrict__ Wt,
    _Float16* __restrict__ Qh, _Float16* __restrict__ Kh,
    _Float16* __restrict__ Vt) {
  const int t0   = blockIdx.x;          // row tile over flattened B*N
  const int wave = threadIdx.x >> 5;
  const int lane = threadIdx.x & 31;
  const int l15  = lane & 15;
  const int hi   = lane >> 4;
  const int d0   = wave * 16;
  const int row0 = t0 * 16;
  const int row  = row0 + l15;

  // A fragments of x (16 rows x 64), chunks c=0 (k 0..31), c=1 (k 32..63)
  v16h a[2];
#pragma unroll
  for (int c = 0; c < 2; ++c) {
    const float* p0 = x + (size_t)row * CIN + c * 32 + hi * 16;
    v4f f0 = *(const v4f*)(p0 + 0),  f1 = *(const v4f*)(p0 + 4);
    v4f f2 = *(const v4f*)(p0 + 8), f3 = *(const v4f*)(p0 + 12);
    v16h t;
#pragma unroll
    for (int i = 0; i < 4; ++i) {
      t[i]      = (_Float16)f0[i];
      t[4 + i]  = (_Float16)f1[i];
      t[8 + i]  = (_Float16)f2[i];
      t[12 + i] = (_Float16)f3[i];
    }
    a[c] = t;
  }

  const int b  = row0 / N_;
  const int nb = row0 % N_;

#pragma unroll
  for (int m = 0; m < 3; ++m) {
    const _Float16* wrow = Wt + ((size_t)m * D_ + d0 + l15) * CIN;
    v16h b0 = *(const v16h*)(wrow + 0  + hi * 16);
    v16h b1 = *(const v16h*)(wrow + 32 + hi * 16);
    v8f acc = {};
    acc = __builtin_amdgcn_wmma_f32_16x16x32_f16(false, a[0], false, b0,
                                                 (short)0, acc, false, false);
    acc = __builtin_amdgcn_wmma_f32_16x16x32_f16(false, a[1], false, b1,
                                                 (short)0, acc, false, false);
    const float* bias = (m == 0) ? bq : ((m == 1) ? bk : bv);
    float bb = bias[d0 + l15];
#pragma unroll
    for (int r = 0; r < 8; ++r) {
      float v = acc[r] + bb;
      v = v > 0.f ? v : 0.f;              // ReLU
      if (m == 0) v *= 0.125f;            // fold 1/sqrt(Cin)
      acc[r] = v;
    }
    if (m == 2) {
      // transposed store: Vt[b][d][n]; 8 consecutive n -> single 16B store
      v8h pk;
#pragma unroll
      for (int r = 0; r < 8; ++r) pk[r] = (_Float16)acc[r];
      _Float16* dst = Vt + ((size_t)b * D_ + d0 + l15) * N_ + nb + 8 * hi;
      *(v8h*)dst = pk;
    } else {
      _Float16* dst = (m == 0) ? Qh : Kh;
#pragma unroll
      for (int r = 0; r < 8; ++r)
        dst[(size_t)(row0 + r + 8 * hi) * D_ + d0 + l15] = (_Float16)acc[r];
    }
  }
}

// ---------------------------------------------------------------------------
// Kernel 2: flash attention, one wave per (batch, 16-query tile).
// 32-key blocks: 8 WMMAs for S, online softmax, LDS relayout of P, 8 WMMAs
// for O += P*V. grid = 256 blocks x 8 waves = 2048 tiles.
// ---------------------------------------------------------------------------
__global__ __launch_bounds__(256) void attn_kernel(
    const _Float16* __restrict__ Qh, const _Float16* __restrict__ Kh,
    const _Float16* __restrict__ Vt, float* __restrict__ out) {
  __shared__ __align__(32) _Float16 plds[8][16][32];   // 1KB per wave
  const int wave = threadIdx.x >> 5;
  const int lane = threadIdx.x & 31;
  const int l15  = lane & 15;
  const int hi   = lane >> 4;
  const int g    = blockIdx.x * 8 + wave;   // tile id in [0, 2048)
  const int b    = g >> 7;                  // / (N/16)
  const int n0   = (g & 127) << 4;

  // Q A-fragments: 4 chunks of 32 along D
  v16h qa[4];
#pragma unroll
  for (int c = 0; c < 4; ++c) {
    const _Float16* p = Qh + (size_t)(b * N_ + n0 + l15) * D_ + c * 32 + hi * 16;
    qa[c] = cat8(*(const v8h*)p, *(const v8h*)(p + 8));
  }

  float m_run[8], l_run[8];
  v8f o[8];
#pragma unroll
  for (int r = 0; r < 8; ++r) { m_run[r] = -3.0e38f; l_run[r] = 0.f; }
#pragma unroll
  for (int t = 0; t < 8; ++t)
#pragma unroll
    for (int r = 0; r < 8; ++r) o[t][r] = 0.f;

  for (int kb = 0; kb < N_; kb += 32) {
    // S = Q * K^T for 32 keys (two 16x16 f32 tiles)
    v8f s_lo = {}, s_hi = {};
#pragma unroll
    for (int c = 0; c < 4; ++c) {
      const _Float16* kp =
          Kh + (size_t)(b * N_ + kb + l15) * D_ + c * 32 + hi * 16;
      v16h kt = *(const v16h*)kp;
      s_lo = __builtin_amdgcn_wmma_f32_16x16x32_f16(false, qa[c], false, kt,
                                                    (short)0, s_lo, false, false);
    }
#pragma unroll
    for (int c = 0; c < 4; ++c) {
      const _Float16* kp =
          Kh + (size_t)(b * N_ + kb + 16 + l15) * D_ + c * 32 + hi * 16;
      v16h kt = *(const v16h*)kp;
      s_hi = __builtin_amdgcn_wmma_f32_16x16x32_f16(false, qa[c], false, kt,
                                                    (short)0, s_hi, false, false);
    }

    // online softmax: rows live as (r, hi) across 16-lane halves
#pragma unroll
    for (int r = 0; r < 8; ++r) {
      float mb = fmaxf(s_lo[r], s_hi[r]);
      mb = fmaxf(mb, __shfl_xor(mb, 1, 32));
      mb = fmaxf(mb, __shfl_xor(mb, 2, 32));
      mb = fmaxf(mb, __shfl_xor(mb, 4, 32));
      mb = fmaxf(mb, __shfl_xor(mb, 8, 32));
      float m_new = fmaxf(m_run[r], mb);
      float fac = __expf(m_run[r] - m_new);
      float pl = __expf(s_lo[r] - m_new);
      float ph = __expf(s_hi[r] - m_new);
      float rs = pl + ph;
      rs += __shfl_xor(rs, 1, 32);
      rs += __shfl_xor(rs, 2, 32);
      rs += __shfl_xor(rs, 4, 32);
      rs += __shfl_xor(rs, 8, 32);
      l_run[r] = l_run[r] * fac + rs;
      m_run[r] = m_new;
#pragma unroll
      for (int t = 0; t < 8; ++t) o[t][r] *= fac;
      // stage P in wave-private LDS: [q][key]
      plds[wave][r + 8 * hi][l15]      = (_Float16)pl;
      plds[wave][r + 8 * hi][16 + l15] = (_Float16)ph;
    }

    // reload P as a 16x32 A-fragment (wave-private LDS, no barrier needed)
    v8h plo = *(const v8h*)&plds[wave][l15][hi * 16];
    v8h phh = *(const v8h*)&plds[wave][l15][hi * 16 + 8];
    v16h pa = cat8(plo, phh);

    // O += P * V  (V transposed: contraction over keys is contiguous)
#pragma unroll
    for (int t = 0; t < 8; ++t) {
      const _Float16* vp =
          Vt + ((size_t)b * D_ + t * 16 + l15) * N_ + kb + hi * 16;
      v16h vt = *(const v16h*)vp;
      o[t] = __builtin_amdgcn_wmma_f32_16x16x32_f16(false, pa, false, vt,
                                                    (short)0, o[t], false, false);
    }
  }

  // epilogue: normalize and store f32
#pragma unroll
  for (int t = 0; t < 8; ++t)
#pragma unroll
    for (int r = 0; r < 8; ++r)
      out[(size_t)(b * N_ + n0 + r + 8 * hi) * D_ + t * 16 + l15] =
          o[t][r] / l_run[r];
}

// ---------------------------------------------------------------------------
extern "C" void kernel_launch(void* const* d_in, const int* in_sizes, int n_in,
                              void* d_out, int out_size, void* d_ws,
                              size_t ws_size, hipStream_t stream) {
  const float* x  = (const float*)d_in[0];
  const float* Wq = (const float*)d_in[1];
  const float* bq = (const float*)d_in[2];
  const float* Wk = (const float*)d_in[3];
  const float* bk = (const float*)d_in[4];
  const float* Wv = (const float*)d_in[5];
  const float* bv = (const float*)d_in[6];
  float* out = (float*)d_out;

  // workspace layout (f16): Wt | Qh | Kh | Vt
  const size_t WT_ELE  = (size_t)3 * D_ * CIN;        // 24576
  const size_t QKV_ELE = (size_t)B_ * N_ * D_;        // 4194304
  _Float16* Wt = (_Float16*)d_ws;
  _Float16* Qh = Wt + WT_ELE;
  _Float16* Kh = Qh + QKV_ELE;
  _Float16* Vt = Kh + QKV_ELE;

  wt_kernel<<<(int)(WT_ELE / 256), 256, 0, stream>>>(Wq, Wk, Wv, Wt);
  qkv_kernel<<<B_ * N_ / 16, 256, 0, stream>>>(x, bq, bk, bv, Wt, Qh, Kh, Vt);
  attn_kernel<<<B_ * N_ / 16 / 8, 256, 0, stream>>>(Qh, Kh, Vt, out);
}
